// Net2_EE_76897094467836
// MI455X (gfx1250) — compile-verified, matrix operand order checked by reading, not run on hardware
//
#include <hip/hip_runtime.h>
#include <stdint.h>

typedef uint32_t u32;
typedef __attribute__((ext_vector_type(16))) _Float16 v16h;
typedef __attribute__((ext_vector_type(8)))  float    v8f;
typedef __attribute__((ext_vector_type(4)))  u32      v4u;
typedef __attribute__((ext_vector_type(8)))  int      v8i;
typedef __attribute__((ext_vector_type(4)))  int      v4i;

// ---------------------------------------------------------------- helpers ---

__device__ __forceinline__ v8f v8f_zero() {
  v8f z = {0.f,0.f,0.f,0.f,0.f,0.f,0.f,0.f};
  return z;
}

__device__ __forceinline__ v8f wmma_f16(v16h a, v16h b, v8f c) {
  // v_wmma_f32_16x16x32_f16  (8-arg form, probe-confirmed)
  return __builtin_amdgcn_wmma_f32_16x16x32_f16(
      /*neg_a=*/false, a, /*neg_b=*/false, b,
      /*c_mod=*/(short)0, c, /*reuse_a=*/false, /*reuse_b=*/false);
}

// Documented 16-bit A/B fragment layout (CDNA5 ISA §7.12.2):
// lane half h (lane>>4) owns k in {8h..8h+7} U {16+8h..16+8h+7}.
__device__ __forceinline__ int frag_k(int e, int hlf) {
  int v = e >> 1, p = e & 1;
  return (v < 4) ? (8*hlf + 2*v + p) : (16 + 8*hlf + 2*(v-4) + p);
}

// Fragment = two 16B chunks from a 64B (32-half) row: bytes [16h,16h+16) and [32+16h,+16)
union FragU { uint4 u[2]; v16h h; };
__device__ __forceinline__ v16h frag_row(const _Float16* row, int hlf) {
  FragU f;
  const char* p = (const char*)row;
  f.u[0] = *(const uint4*)(p + 16*hlf);
  f.u[1] = *(const uint4*)(p + 32 + 16*hlf);
  return f.h;
}

__device__ __forceinline__ u32 lds_addr32(const void* p) {
  // flat shared aperture keeps LDS byte offset in addr[31:0]
  return (u32)(uintptr_t)p;
}

// Tensor Data Mover: 2D f16 tile (tile1 rows x tile0 contiguous elems) -> LDS.
// 6-arg builtin form (amdgpu-toolchain / clang-23).  D# packing per
// CDNA5 ISA: group0 = control/lds/global addr,
// group1 = dims/tile/strides, groups 2/3 (and trailing group) zero for 2D tiles.
__device__ __forceinline__ void tdm_load_2d_f16(u32 lds_byte, const void* gptr,
                                                u32 td0, u32 td1,
                                                u32 tile0, u32 tile1, u32 stride0)
{
  unsigned long long ga = (unsigned long long)gptr;
  v4u g0;
  g0[0] = 1u;                                             // count=1, user descriptor
  g0[1] = lds_byte;                                       // lds_addr
  g0[2] = (u32)ga;                                        // global_addr[31:0]
  g0[3] = (u32)((ga >> 32) & 0x1FFFFFFu) | 0x80000000u;   // addr[56:32] | type=2
  v8i g1;
  g1[0] = (int)(1u << 16);                                // wg_mask=0, data_size=2B
  g1[1] = (int)((td0 & 0xFFFFu) << 16);                   // tensor_dim0[15:0]
  g1[2] = (int)(((td0 >> 16) & 0xFFFFu) | ((td1 & 0xFFFFu) << 16));
  g1[3] = (int)(((td1 >> 16) & 0xFFFFu) | ((tile0 & 0xFFFFu) << 16));
  g1[4] = (int)(tile1 & 0xFFFFu);                         // tile_dim1, tile_dim2=0
  g1[5] = (int)stride0;                                   // tensor_dim0_stride[31:0]
  g1[6] = 0;                                              // stride hi | dim1_stride lo
  g1[7] = 0;
  v4i gz4 = {0,0,0,0};
  v8i gz8 = {0,0,0,0,0,0,0,0};
  __builtin_amdgcn_tensor_load_to_lds(g0, g1, gz4, gz4, gz8, 0);
}

// ------------------------------------------------------------ constants -----

#define B_IMG   4096
#define LOW_T   (60.0f/255.0f)
#define HIGH_T  (120.0f/255.0f)

// ------------------------------------------------ kernel 0: weight casts ----

__global__ void k_convert(const float* __restrict__ conv2_w,
                          const float* __restrict__ fc1_w,
                          _Float16* __restrict__ w2h,   // [64][800] (=[n][k])
                          _Float16* __restrict__ w1h)   // [1024][1024] (=[n][k])
{
  const int N1 = 64*800, N2 = 1024*1024;
  for (int idx = blockIdx.x*blockDim.x + threadIdx.x; idx < N1+N2;
       idx += gridDim.x*blockDim.x) {
    if (idx < N1) w2h[idx]      = (_Float16)conv2_w[idx];
    else          w1h[idx - N1] = (_Float16)fc1_w[idx - N1];
  }
}

// --------------------------- kernel 1: hfs + canny + conv1 + pool (f32) -----

__global__ void __launch_bounds__(256)
k_pre_conv1(const float* __restrict__ x, const float* __restrict__ w1,
            const float* __restrict__ b1, _Float16* __restrict__ act1)
{
  __shared__ float s_img[784], s_yr[784], s_yi[784], s_a[784], s_b[784];
  __shared__ float s_cr[28], s_ci[28];
  __shared__ float s_w[800];
  const int t = threadIdx.x;
  const int bimg = blockIdx.x;
  const float* xi = x + (size_t)bimg*784;

  for (int i = t; i < 784; i += 256) s_img[i] = xi[i];
  for (int i = t; i < 800; i += 256) s_w[i]   = w1[i];

  // circulant kernels of the low-pass operator: kept freqs f with (f+14)%28 in [6,22)
  if (t < 28) {
    float cr = 0.f, ci = 0.f;
    for (int f = 0; f < 28; ++f) {
      int cidx = (f + 14) % 28;
      if (cidx >= 6 && cidx < 22) {
        float th = 6.28318530717958647f * (float)((f * t) % 28) * (1.0f/28.0f);
        float sn, cs; __sincosf(th, &sn, &cs);
        cr += cs; ci += sn;
      }
    }
    s_cr[t] = cr * (1.0f/28.0f);
    s_ci[t] = ci * (1.0f/28.0f);
  }
  __syncthreads();

  // Y = A_c @ X (complex rows)
  for (int p = t; p < 784; p += 256) {
    int i = p / 28, j = p - 28*(p/28);
    float yr = 0.f, yi = 0.f;
    for (int k = 0; k < 28; ++k) {
      int d = i - k; d += (d < 0) ? 28 : 0;
      float v = s_img[k*28 + j];
      yr += s_cr[d] * v;
      yi += s_ci[d] * v;
    }
    s_yr[p] = yr; s_yi[p] = yi;
  }
  __syncthreads();
  // hfs = Re(Y @ A_c^T) -> s_a
  for (int p = t; p < 784; p += 256) {
    int i = p / 28, j = p % 28;
    float z = 0.f;
    for (int k = 0; k < 28; ++k) {
      int d = j - k; d += (d < 0) ? 28 : 0;
      z += s_yr[i*28 + k] * s_cr[d] - s_yi[i*28 + k] * s_ci[d];
    }
    s_a[p] = z;
  }
  __syncthreads();

  // 3x3 gaussian (sigma=1), zero-pad SAME -> s_b
  {
    const float e1 = 0.60653065971f, e2 = 0.36787944117f;
    const float gn = 1.0f / (1.0f + 4.0f*e1 + 4.0f*e2);
    const float GC = gn, GE = e1*gn, GK = e2*gn;
    for (int p = t; p < 784; p += 256) {
      int y = p / 28, xx = p % 28;
      float acc = 0.f;
      #pragma unroll
      for (int dy = -1; dy <= 1; ++dy)
        #pragma unroll
        for (int dx = -1; dx <= 1; ++dx) {
          int yy = y+dy, xc = xx+dx;
          if ((unsigned)yy < 28u && (unsigned)xc < 28u) {
            int ad = dy*dy + dx*dx;
            float gw = (ad==0) ? GC : ((ad==1) ? GE : GK);
            acc += gw * s_img[yy*28 + xc];
          }
        }
      s_b[p] = acc;
    }
  }
  __syncthreads();

  // custom sobel (x/(x^2+y^2)) -> gx:s_yr gy:s_yi
  for (int p = t; p < 784; p += 256) {
    int y = p / 28, xx = p % 28;
    auto B = [&](int yy, int xc) -> float {
      return ((unsigned)yy < 28u && (unsigned)xc < 28u) ? s_b[yy*28 + xc] : 0.f;
    };
    float gx = 0.5f*(B(y-1,xx+1)-B(y-1,xx-1)) + (B(y,xx+1)-B(y,xx-1))
             + 0.5f*(B(y+1,xx+1)-B(y+1,xx-1));
    float gy = 0.5f*(B(y+1,xx-1)-B(y-1,xx-1)) + (B(y+1,xx)-B(y-1,xx))
             + 0.5f*(B(y+1,xx+1)-B(y-1,xx+1));
    s_yr[p] = gx; s_yi[p] = gy;
  }
  __syncthreads();
  // mag -> s_b, orientation bucket -> s_img
  for (int p = t; p < 784; p += 256) {
    float gx = s_yr[p], gy = s_yi[p];
    float mag = sqrtf(gx*gx + gy*gy + 1e-12f);
    float ori = atan2f(gy, gx) * 57.2957795130823f + 180.0f;
    int q = (int)roundf(ori * (1.0f/45.0f));
    s_b[p]  = mag;
    s_img[p] = (float)(q & 7);
  }
  __syncthreads();
  // NMS -> thin in s_yr
  {
    const int offr[8] = {0,-1,-1,-1, 0, 1, 1, 1};
    const int offc[8] = {1, 1, 0,-1,-1,-1, 0, 1};
    for (int p = t; p < 784; p += 256) {
      int y = p / 28, xx = p % 28;
      float m = s_b[p];
      int pos = (int)s_img[p];
      int ax = pos & 3;
      auto M = [&](int dr, int dc) -> float {
        int yy = y+dr, xc = xx+dc;
        return ((unsigned)yy < 28u && (unsigned)xc < 28u) ? s_b[yy*28 + xc] : 0.f;
      };
      float d1 = m - M(offr[ax],   offc[ax]);
      float d2 = m - M(offr[ax+4], offc[ax+4]);
      s_yr[p] = (fminf(d1, d2) > 0.f) ? m : 0.f;
    }
  }
  __syncthreads();
  // thresholds: strong->s_yi  weak->s_img
  for (int p = t; p < 784; p += 256) {
    float th = s_yr[p];
    s_yi[p]  = (th > HIGH_T) ? 1.f : 0.f;
    s_img[p] = (th > LOW_T && th <= HIGH_T) ? 1.f : 0.f;
  }
  __syncthreads();
  // 10 iterations of masked dilation (hysteresis), ping-pong s_yi <-> s_b
  {
    float* S = s_yi; float* T = s_b;
    for (int it = 0; it < 10; ++it) {
      for (int p = t; p < 784; p += 256) {
        int y = p / 28, xx = p % 28;
        float any = 0.f;
        #pragma unroll
        for (int dy = -1; dy <= 1; ++dy)
          #pragma unroll
          for (int dx = -1; dx <= 1; ++dx) {
            int yy = y+dy, xc = xx+dx;
            if ((unsigned)yy < 28u && (unsigned)xc < 28u) any = fmaxf(any, S[yy*28+xc]);
          }
        T[p] = fmaxf(S[p], (any > 0.f) ? s_img[p] : 0.f);
      }
      __syncthreads();
      float* tmp = S; S = T; T = tmp;
    }
    // h = clip(hfs + strong, 0, 1) -> s_a
    for (int p = t; p < 784; p += 256)
      s_a[p] = fminf(fmaxf(s_a[p] + S[p], 0.f), 1.f);
  }
  __syncthreads();

  // conv1 (5x5 VALID, 32 oc) fused with 2x2 maxpool + relu -> act1[b][oc][12][12] f16
  for (int idx = t; idx < 4608; idx += 256) {
    int oc = idx / 144, p = idx - oc*144;
    int py = p / 12, px = p - py*12;
    float mx = -1e30f;
    #pragma unroll
    for (int sy = 0; sy < 2; ++sy)
      #pragma unroll
      for (int sx = 0; sx < 2; ++sx) {
        int oy = 2*py + sy, ox = 2*px + sx;
        float acc = b1[oc];
        #pragma unroll
        for (int ky = 0; ky < 5; ++ky)
          #pragma unroll
          for (int kx = 0; kx < 5; ++kx)
            acc += s_a[(oy+ky)*28 + (ox+kx)] * s_w[oc*25 + ky*5 + kx];
        mx = fmaxf(mx, acc);
      }
    act1[(size_t)bimg*4608 + idx] = (_Float16)fmaxf(mx, 0.f);
  }
}

// ------------- kernel 2: conv2 implicit-GEMM WMMA + fused pool/relu ---------
// 4 images/block.  M=4*64 positions, N=64 oc, K=800=25*32.  Weights via TDM.

__global__ void __launch_bounds__(256)
k_conv2(const _Float16* __restrict__ act1, const _Float16* __restrict__ w2h,
        const float* __restrict__ b2, _Float16* __restrict__ h2)
{
  __shared__ __align__(16) _Float16 s_act[4*4608];      // also reused for results
  __shared__ __align__(16) _Float16 s_w[2][64*32];      // [n][k] tile, double buffered
  const int t = threadIdx.x;
  const int lane = t & 31, l15 = lane & 15, hlf = (lane >> 4) & 1, wv = t >> 5;

  // stage 4 images of activations (b128 loads)
  {
    const uint4* ga = (const uint4*)(act1 + (size_t)blockIdx.x*4*4608);
    uint4* sa = (uint4*)s_act;
    for (int i = t; i < 2304; i += 256) sa[i] = ga[i];
  }
  // TDM: first weight K-slab  (tile 64 rows x 32 k, row stride 800)
  if (t < 32) {
    tdm_load_2d_f16(lds_addr32(&s_w[0][0]), (const void*)w2h, 800, 64, 32, 64, 800);
    __builtin_amdgcn_s_wait_tensorcnt(0);
  }
  __syncthreads();

  v8f acc[2][4];
  #pragma unroll
  for (int a = 0; a < 2; ++a)
    #pragma unroll
    for (int nb = 0; nb < 4; ++nb) acc[a][nb] = v8f_zero();

  for (int kb = 0; kb < 25; ++kb) {
    const int cur = kb & 1;
    if (kb + 1 < 25 && t < 32)
      tdm_load_2d_f16(lds_addr32(&s_w[cur ^ 1][0]), (const void*)(w2h + (kb+1)*32),
                      800, 64, 32, 64, 800);
    // A fragments: im2col gather from LDS
    v16h afr[2];
    #pragma unroll
    for (int a = 0; a < 2; ++a) {
      int mt = (a == 0) ? wv : (wv + 8);
      int im = mt >> 2;
      int m  = ((mt & 3) << 4) + l15;
      int oy = m >> 3, ox = m & 7;
      const _Float16* actim = s_act + im*4608;
      v16h av;
      #pragma unroll
      for (int e = 0; e < 16; ++e) {
        int k  = kb*32 + frag_k(e, hlf);
        int ic = k / 25, r = k - ic*25;
        int ky = r / 5,  kx = r - ky*5;
        av[e] = actim[ic*144 + (oy+ky)*12 + (ox+kx)];
      }
      afr[a] = av;
    }
    // B fragments (contiguous 128-bit row chunks) + 8 WMMAs
    #pragma unroll
    for (int nb = 0; nb < 4; ++nb) {
      int n = nb*16 + l15;
      v16h bv = frag_row(&s_w[cur][n*32], hlf);
      acc[0][nb] = wmma_f16(afr[0], bv, acc[0][nb]);
      acc[1][nb] = wmma_f16(afr[1], bv, acc[1][nb]);
    }
    if (t < 32) __builtin_amdgcn_s_wait_tensorcnt(0);
    __syncthreads();
  }

  // spill results (f16) into reused activation LDS: res[im][pos][oc]
  _Float16* s_res = s_act;
  #pragma unroll
  for (int a = 0; a < 2; ++a) {
    int mt = (a == 0) ? wv : (wv + 8);
    int im = mt >> 2;
    int mb = (mt & 3) << 4;
    #pragma unroll
    for (int nb = 0; nb < 4; ++nb)
      #pragma unroll
      for (int r = 0; r < 8; ++r) {
        int m = mb + r + 8*hlf;
        int n = nb*16 + l15;
        s_res[im*4096 + m*64 + n] = (_Float16)acc[a][nb][r];
      }
  }
  __syncthreads();

  // 2x2 maxpool + bias + relu -> h2[b][oc*16+py*4+px]
  for (int idx = t; idx < 4096; idx += 256) {
    int im = idx >> 10;
    int f  = idx & 1023;
    int oc = f >> 4;
    int py = (f >> 2) & 3, px = f & 3;
    const _Float16* rim = s_res + im*4096;
    float v = -1e30f;
    #pragma unroll
    for (int sy = 0; sy < 2; ++sy)
      #pragma unroll
      for (int sx = 0; sx < 2; ++sx) {
        int pos = (2*py+sy)*8 + (2*px+sx);
        v = fmaxf(v, (float)rim[pos*64 + oc]);
      }
    v = fmaxf(v + b2[oc], 0.f);
    h2[(size_t)(blockIdx.x*4 + im)*1024 + f] = (_Float16)v;
  }
}

// -------------------- kernel 3: fc1 WMMA GEMM (TDM double-buffered) ---------
// 64x64 block tile, 4 waves each 32x32.  A = h2[4096][1024], B kept [n][k].

__global__ void __launch_bounds__(128)
k_fc1(const _Float16* __restrict__ h2, const _Float16* __restrict__ w1h,
      const float* __restrict__ fb1, _Float16* __restrict__ h3)
{
  __shared__ __align__(16) _Float16 As[2][64*32];
  __shared__ __align__(16) _Float16 Bs[2][64*32];
  const int t = threadIdx.x;
  const int lane = t & 31, l15 = lane & 15, hlf = (lane >> 4) & 1, wv = t >> 5;
  const int nt = blockIdx.x & 15, mt = blockIdx.x >> 4;
  const int m0 = mt*64, n0 = nt*64;
  const int wm = (wv >> 1)*32, wn = (wv & 1)*32;

  if (t < 32) {
    tdm_load_2d_f16(lds_addr32(&As[0][0]), (const void*)(h2 + (size_t)m0*1024),
                    1024, 4096, 32, 64, 1024);
    tdm_load_2d_f16(lds_addr32(&Bs[0][0]), (const void*)(w1h + (size_t)n0*1024),
                    1024, 1024, 32, 64, 1024);
    __builtin_amdgcn_s_wait_tensorcnt(0);
  }
  __syncthreads();

  v8f acc[2][2];
  #pragma unroll
  for (int a = 0; a < 2; ++a)
    #pragma unroll
    for (int b = 0; b < 2; ++b) acc[a][b] = v8f_zero();

  for (int kb = 0; kb < 32; ++kb) {
    const int cur = kb & 1;
    if (kb + 1 < 32 && t < 32) {
      tdm_load_2d_f16(lds_addr32(&As[cur ^ 1][0]),
                      (const void*)(h2 + (size_t)m0*1024 + (kb+1)*32),
                      1024, 4096, 32, 64, 1024);
      tdm_load_2d_f16(lds_addr32(&Bs[cur ^ 1][0]),
                      (const void*)(w1h + (size_t)n0*1024 + (kb+1)*32),
                      1024, 1024, 32, 64, 1024);
    }
    v16h af[2], bf[2];
    #pragma unroll
    for (int a = 0; a < 2; ++a)
      af[a] = frag_row(&As[cur][(wm + a*16 + l15)*32], hlf);
    #pragma unroll
    for (int b = 0; b < 2; ++b)
      bf[b] = frag_row(&Bs[cur][(wn + b*16 + l15)*32], hlf);
    #pragma unroll
    for (int a = 0; a < 2; ++a)
      #pragma unroll
      for (int b = 0; b < 2; ++b)
        acc[a][b] = wmma_f16(af[a], bf[b], acc[a][b]);
    if (t < 32) __builtin_amdgcn_s_wait_tensorcnt(0);
    __syncthreads();
  }

  // epilogue: +bias, relu, cast f16
  #pragma unroll
  for (int a = 0; a < 2; ++a)
    #pragma unroll
    for (int b = 0; b < 2; ++b)
      #pragma unroll
      for (int r = 0; r < 8; ++r) {
        int m = m0 + wm + a*16 + r + 8*hlf;
        int n = n0 + wn + b*16 + l15;
        h3[(size_t)m*1024 + n] = (_Float16)fmaxf(acc[a][b][r] + fb1[n], 0.f);
      }
}

// ------------------------------ kernel 4: fc2 (N=10, fp32) ------------------

__global__ void k_fc2(const _Float16* __restrict__ h3, const float* __restrict__ w2,
                      const float* __restrict__ b2, float* __restrict__ out)
{
  int idx = blockIdx.x*blockDim.x + threadIdx.x;
  if (idx >= B_IMG*10) return;
  int b = idx / 10, o = idx - b*10;
  const _Float16* hr = h3 + (size_t)b*1024;
  const float*    wr = w2 + o*1024;
  float acc = 0.f;
  #pragma unroll 8
  for (int k = 0; k < 1024; ++k) acc += (float)hr[k] * wr[k];
  out[idx] = acc + b2[o];
}

// ------------------------------------------------------------- launcher -----

extern "C" void kernel_launch(void* const* d_in, const int* in_sizes, int n_in,
                              void* d_out, int out_size, void* d_ws, size_t ws_size,
                              hipStream_t stream) {
  const float* x   = (const float*)d_in[0];
  const float* c1w = (const float*)d_in[1];
  const float* c1b = (const float*)d_in[2];
  const float* c2w = (const float*)d_in[3];
  const float* c2b = (const float*)d_in[4];
  const float* f1w = (const float*)d_in[5];
  const float* f1b = (const float*)d_in[6];
  const float* f2w = (const float*)d_in[7];
  const float* f2b = (const float*)d_in[8];
  (void)in_sizes; (void)n_in; (void)out_size; (void)ws_size;

  char* ws = (char*)d_ws;
  _Float16* w2h  = (_Float16*)(ws);                 //   102,400 B
  _Float16* w1h  = (_Float16*)(ws + 102400);        // 2,097,152 B
  _Float16* act1 = (_Float16*)(ws + 2199552);       // 37,748,736 B
  _Float16* h2   = (_Float16*)(ws + 39948288);      // 8,388,608 B
  _Float16* h3   = (_Float16*)(ws + 48336896);      // 8,388,608 B  (total ~54 MB)
  float* out = (float*)d_out;

  k_convert  <<<512,  256, 0, stream>>>(c2w, f1w, w2h, w1h);
  k_pre_conv1<<<4096, 256, 0, stream>>>(x, c1w, c1b, act1);
  k_conv2    <<<1024, 256, 0, stream>>>(act1, w2h, c2b, h2);
  k_fc1      <<<1024, 128, 0, stream>>>(h2, w1h, f1b, h3);
  k_fc2      <<<(B_IMG*10 + 255)/256, 256, 0, stream>>>(h3, f2w, f2b, out);
}
